// CrossAttnBlockppTwoCams_78288663871978
// MI455X (gfx1250) — compile-verified, matrix operand order checked by reading, not run on hardware
//
#include <hip/hip_runtime.h>
#include <math.h>
#include <stdint.h>

// ---------------------------------------------------------------------------
// Problem constants
// ---------------------------------------------------------------------------
#define B_    16
#define C_    128
#define HW_   1024         // 32*32
#define NH_   4
#define COND_ 32
#define FAN0  160          // C + COND
#define NQK   512          // C*NH
#define FAN3  768          // (NH+2)*C
#define EPS_  1e-6f
#define SCALE_ 0.08838834764831845f   // 128^-0.5

typedef __attribute__((ext_vector_type(16))) _Float16 v16h;
typedef __attribute__((ext_vector_type(8)))  _Float16 v8h;
typedef __attribute__((ext_vector_type(8)))  float    v8f;
typedef int v4i __attribute__((vector_size(16)));   // matches async-LDS builtin param

// Workspace layout (units: _Float16 elements). Total ~86.7 MB.
static constexpr size_t SZ_A   = (size_t)B_ * HW_ * FAN0;
static constexpr size_t SZ_QKV = (size_t)B_ * NH_ * HW_ * C_;
static constexpr size_t SZ_HH  = (size_t)B_ * HW_ * FAN3;
static constexpr size_t SZ_W   = (size_t)FAN0 * NQK;
static constexpr size_t SZ_W3  = (size_t)FAN3 * C_;

static constexpr size_t OFF_AQ = 0;                  // [b][hw][160] f16
static constexpr size_t OFF_AK = OFF_AQ + SZ_A;
static constexpr size_t OFF_Q  = OFF_AK + SZ_A;      // [b][e][q][c]  f16
static constexpr size_t OFF_K  = OFF_Q  + SZ_QKV;    // [b][e][k][c]  f16
static constexpr size_t OFF_V  = OFF_K  + SZ_QKV;    // [b][e][c][k]  f16
static constexpr size_t OFF_HH = OFF_V  + SZ_QKV;    // [b'][q][768]  f16
static constexpr size_t OFF_W0 = OFF_HH + SZ_HH;     // transposed [512][160]
static constexpr size_t OFF_W1 = OFF_W0 + SZ_W;
static constexpr size_t OFF_W2 = OFF_W1 + SZ_W;
static constexpr size_t OFF_W3 = OFF_W2 + SZ_W;      // transposed [128][768]

// ---------------------------------------------------------------------------
// CDNA5 async global->LDS copy (16B per lane), ASYNCcnt-tracked.
// Builtin when the toolchain declares it, else inline gfx1250 asm.
// ---------------------------------------------------------------------------
#if __has_builtin(__builtin_amdgcn_global_load_async_to_lds_b128)
#define ASYNC_MODE 2
#elif defined(__AMDGCN__)
#define ASYNC_MODE 1
#else
#define ASYNC_MODE 0
#endif

__device__ __forceinline__ void cp16_g2l(_Float16* ldst, const _Float16* gsrc) {
#if ASYNC_MODE == 2
  __builtin_amdgcn_global_load_async_to_lds_b128((v4i*)gsrc, (v4i*)ldst, 0, 0);
#elif ASYNC_MODE == 1
  unsigned l = (unsigned)(uintptr_t)ldst;           // LDS byte address
  unsigned long long g = (unsigned long long)(uintptr_t)gsrc;
  asm volatile("global_load_async_to_lds_b128 %0, %1, off"
               :: "v"(l), "v"(g) : "memory");
#else
  *(v8h*)ldst = *(const v8h*)gsrc;                  // host-pass placeholder
#endif
}

__device__ __forceinline__ void wait_async() {
#if ASYNC_MODE == 0
#elif __has_builtin(__builtin_amdgcn_s_wait_asynccnt)
  __builtin_amdgcn_s_wait_asynccnt(0);
#else
  asm volatile("s_wait_asynccnt 0x0" ::: "memory");
#endif
}

// ---------------------------------------------------------------------------
// Fragment loader.  ISA 16-bit A 16x32 layout: lanes 0-15 = rows, halves
// 0-7 -> K 0..7, halves 8-15 -> K 16..23; lanes 16-31: K 8..15 / 24..31.
// B 32x16 is the mirror (lane = column).  Both reduce to:
//   sel = (lane&16)?8:0 ; lo = rowp[sel..sel+7] ; hi = rowp[sel+16..sel+23]
// ---------------------------------------------------------------------------
__device__ __forceinline__ v16h frag16(const _Float16* rowp, int lane) {
  const int sel = (lane & 16) ? 8 : 0;
  v8h lo = *(const v8h*)(rowp + sel);
  v8h hi = *(const v8h*)(rowp + sel + 16);
  return __builtin_shufflevector(lo, hi, 0,1,2,3,4,5,6,7,8,9,10,11,12,13,14,15);
}

__device__ __forceinline__ v8f wmma_f16(v16h a, v16h b, v8f c) {
  return __builtin_amdgcn_wmma_f32_16x16x32_f16(false, a, false, b,
                                                (short)0, c, false, false);
}

__device__ __forceinline__ v8f v8f_zero() {
  v8f z; for (int i = 0; i < 8; ++i) z[i] = 0.0f; return z;
}

// ---------------------------------------------------------------------------
// Kernel 0: weights f32 -> f16, pre-transposed to [n][k] so GEMM B staging
// is a contiguous 16B block copy.
// ---------------------------------------------------------------------------
__global__ void k_wcvt(const float* __restrict__ W0, const float* __restrict__ W1,
                       const float* __restrict__ W2, const float* __restrict__ W3,
                       _Float16* __restrict__ ws) {
  size_t i = (size_t)blockIdx.x * 256 + threadIdx.x;
  if (i < SZ_W) {                      // W* row-major [160][512] -> [512][160]
    int n = (int)(i & 511), k = (int)(i >> 9);
    size_t o = (size_t)n * FAN0 + k;
    ws[OFF_W0 + o] = (_Float16)W0[i];
    ws[OFF_W1 + o] = (_Float16)W1[i];
    ws[OFF_W2 + o] = (_Float16)W2[i];
  }
  if (i < SZ_W3) {                     // W3 [768][128] -> [128][768]
    int n = (int)(i & 127), k = (int)(i >> 7);
    ws[OFF_W3 + (size_t)n * FAN3 + k] = (_Float16)W3[i];
  }
}

// ---------------------------------------------------------------------------
// Kernel 1: cond columns (128..159) of A_q / A_k.
// q_cs = repeat(q_cond,2); k_cs = interleave(k_a, k_b)
// ---------------------------------------------------------------------------
__global__ void k_cond(const float* __restrict__ qc, const float* __restrict__ ka,
                       const float* __restrict__ kb, _Float16* __restrict__ ws) {
  int i  = blockIdx.x * 256 + threadIdx.x;     // over B*COND*HW = 524288
  int hw = i & (HW_ - 1);
  int cc = (i >> 10) & (COND_ - 1);
  int b  = i >> 15;
  size_t src = ((size_t)(b >> 1) * COND_ + cc) * HW_ + hw;
  const float* ksrc = (b & 1) ? kb : ka;
  size_t row = ((size_t)b * HW_ + hw) * FAN0 + C_ + cc;
  ws[OFF_AQ + row] = (_Float16)qc[src];
  ws[OFF_AK + row] = (_Float16)ksrc[src];
}

// ---------------------------------------------------------------------------
// Kernel 2: GroupNorm -> h (f16) into cols 0..127 of A_q and A_k.
// one block per (batch, group); group = 4 ch x 1024 spatial
// ---------------------------------------------------------------------------
__global__ void k_gnorm(const float* __restrict__ x, const float* __restrict__ sc,
                        const float* __restrict__ bs, _Float16* __restrict__ ws) {
  int b = blockIdx.x >> 5;
  int g = blockIdx.x & 31;
  int t = threadIdx.x;
  __shared__ float r1[256], r2[256];
  const float* xp = x + ((size_t)b * C_ + g * 4) * HW_;
  float vals[16], s = 0.f, ss = 0.f;
  for (int i = 0; i < 16; ++i) {
    int idx = i * 256 + t;
    float v = xp[idx];
    vals[i] = v; s += v; ss += v * v;
  }
  r1[t] = s; r2[t] = ss;
  __syncthreads();
  for (int off = 128; off > 0; off >>= 1) {
    if (t < off) { r1[t] += r1[t + off]; r2[t] += r2[t + off]; }
    __syncthreads();
  }
  float mu   = r1[0] * (1.0f / 4096.0f);
  float var  = r2[0] * (1.0f / 4096.0f) - mu * mu;
  float rinv = rsqrtf(var + EPS_);
  for (int i = 0; i < 16; ++i) {
    int idx = i * 256 + t;
    int ch  = g * 4 + (idx >> 10);
    int hw  = idx & (HW_ - 1);
    float y = (vals[i] - mu) * rinv * sc[ch] + bs[ch];
    _Float16 h = (_Float16)y;
    size_t row = ((size_t)b * HW_ + hw) * FAN0 + ch;
    ws[OFF_AQ + row] = h;
    ws[OFF_AK + row] = h;
  }
}

// ---------------------------------------------------------------------------
// Kernel 3: QKV projection GEMM  (out[1024,512] = A[1024,160] x W + bias).
// 8 waves, 128x128 tile, K = 5 x 32, double-buffered async LDS staging.
// Epilogue scatters into attention layouts, folding 1/sqrt(C) into Q:
//   Q -> [q][c],  K -> [k][c],  V -> [c][k]
// ---------------------------------------------------------------------------
__global__ void k_qkv(const float* __restrict__ b0, const float* __restrict__ b1,
                      const float* __restrict__ b2, _Float16* __restrict__ ws) {
  const int mblk  = blockIdx.x;          // 0..7
  const int nblk  = blockIdx.y;          // 0..3
  const int b     = blockIdx.z / 3;
  const int which = blockIdx.z % 3;
  const int t = threadIdx.x, w = t >> 5, lane = t & 31;

  const _Float16* A  = ws + (which == 0 ? OFF_AQ : OFF_AK) + (size_t)b * HW_ * FAN0;
  const _Float16* Wm = ws + (which == 0 ? OFF_W0 : (which == 1 ? OFF_W1 : OFF_W2));
  const float* bias  = (which == 0) ? b0 : (which == 1 ? b1 : b2);

  __shared__ alignas(16) _Float16 Asub[2][128 * 32];   // [m][k]
  __shared__ alignas(16) _Float16 Bsub[2][128 * 32];   // [n][k]

  const int m0 = mblk * 128, n0 = nblk * 128;

  auto stage = [&](int buf, int k0) {
    for (int i = 0; i < 2; ++i) {
      int idx = t + 256 * i;                    // 0..511 : 128 rows x 4 chunks
      int row = idx >> 2, part = (idx & 3) * 8;
      cp16_g2l(&Asub[buf][row * 32 + part], A  + (size_t)(m0 + row) * FAN0 + k0 + part);
      cp16_g2l(&Bsub[buf][row * 32 + part], Wm + (size_t)(n0 + row) * FAN0 + k0 + part);
    }
  };

  v8f acc[8];
  for (int i = 0; i < 8; ++i) acc[i] = v8f_zero();
  stage(0, 0);

  for (int ks = 0; ks < 5; ++ks) {
    wait_async();
    __syncthreads();
    if (ks + 1 < 5) stage((ks + 1) & 1, (ks + 1) * 32);
    const _Float16* As = Asub[ks & 1];
    const _Float16* Bs = Bsub[ks & 1];
    v16h af = frag16(&As[(w * 16 + (lane & 15)) * 32], lane);
    for (int nt = 0; nt < 8; ++nt) {
      v16h bf = frag16(&Bs[(nt * 16 + (lane & 15)) * 32], lane);
      acc[nt] = wmma_f16(af, bf, acc[nt]);
    }
    __syncthreads();
  }

  const int rbase = m0 + w * 16 + ((lane & 16) ? 8 : 0);
  for (int nt = 0; nt < 8; ++nt) {
    int colg = n0 + nt * 16 + (lane & 15);
    float bv = bias[colg];
    int e = colg >> 7, c = colg & 127;
    size_t be = (size_t)(b * NH_ + e);
    for (int j = 0; j < 8; ++j) {
      int row = rbase + j;
      float val = acc[nt][j] + bv;
      if (which == 0) val *= SCALE_;
      _Float16 h = (_Float16)val;
      if (which == 2)      ws[OFF_V + (be * C_ + c) * HW_ + row] = h;   // [c][k]
      else if (which == 0) ws[OFF_Q + (be * HW_ + row) * C_ + c] = h;   // [q][c]
      else                 ws[OFF_K + (be * HW_ + row) * C_ + c] = h;   // [k][c]
    }
  }
}

// ---------------------------------------------------------------------------
// Kernel 4: flash attention.  96 instances x 8 q-blocks; 8 waves per block,
// each wave owns a 16-row q tile (Q fragments register-resident) and the full
// [16 x 128] O accumulator.  32-wide K chunks, double-buffered async staging.
// ---------------------------------------------------------------------------
__global__ void __launch_bounds__(256, 1)
k_attn(_Float16* __restrict__ ws) {
  const int qb   = blockIdx.x;            // 0..7
  const int inst = blockIdx.y;            // 0..95
  const int p = inst >> 4, g = (inst >> 2) & 3, e = inst & 3;
  // pairing tables: q offsets 0,1,2,3,0,2  /  kv offsets 1,0,3,2,2,0
  const int bq  = 4 * g + ((0x203210 >> (4 * p)) & 0xF);
  const int bkv = 4 * g + ((0x022301 >> (4 * p)) & 0xF);
  const int colt = p * 512 + e * 128;
  const int bo   = g * 4 + colt / FAN3;
  const int col768 = colt % FAN3;

  const int t = threadIdx.x, w = t >> 5, lane = t & 31;
  const _Float16* Qp = ws + OFF_Q + (size_t)(bq  * NH_ + e) * HW_ * C_;  // [q][c]
  const _Float16* Kp = ws + OFF_K + (size_t)(bkv * NH_ + e) * HW_ * C_;  // [k][c]
  const _Float16* Vp = ws + OFF_V + (size_t)(bkv * NH_ + e) * C_ * HW_;  // [c][k]

  __shared__ alignas(16) _Float16 sK[2][32 * 128];   // [kpos][c]
  __shared__ alignas(16) _Float16 sV[2][128 * 32];   // [c][kpos]
  __shared__ alignas(16) _Float16 sP[8 * 16 * 32];   // per-wave P transpose

  auto stage = [&](int buf, int k0) {
    for (int i = 0; i < 2; ++i) {                    // K chunk: contiguous 8KB
      int idx = t + 256 * i;
      cp16_g2l(&sK[buf][idx * 8], Kp + (size_t)k0 * C_ + idx * 8);
    }
    for (int i = 0; i < 2; ++i) {                    // V chunk: 128 rows x 64B
      int idx = t + 256 * i;
      int c = idx >> 2, part = (idx & 3) * 8;
      cp16_g2l(&sV[buf][c * 32 + part], Vp + (size_t)c * HW_ + k0 + part);
    }
  };

  // Q fragments (row = q position, K-dim = channels), register resident
  const int qrow = qb * 128 + w * 16 + (lane & 15);
  v16h qf[4];
  for (int j = 0; j < 4; ++j) qf[j] = frag16(Qp + (size_t)qrow * C_ + 32 * j, lane);

  v8f acc[8];
  for (int i = 0; i < 8; ++i) acc[i] = v8f_zero();
  float m[8], l[8];
  for (int j = 0; j < 8; ++j) { m[j] = -1e30f; l[j] = 0.0f; }

  stage(0, 0);

  for (int kc = 0; kc < 32; ++kc) {
    wait_async();
    __syncthreads();
    if (kc + 1 < 32) stage((kc + 1) & 1, (kc + 1) * 32);
    const _Float16* bK = sK[kc & 1];
    const _Float16* bV = sV[kc & 1];

    // S = Q^T K for two 16-wide k sub-tiles (scale pre-folded into Q)
    v8f s0 = v8f_zero(), s1 = v8f_zero();
    for (int j = 0; j < 4; ++j) {
      v16h bf0 = frag16(&bK[((lane & 15)) * 128 + 32 * j], lane);
      v16h bf1 = frag16(&bK[(16 + (lane & 15)) * 128 + 32 * j], lane);
      s0 = wmma_f16(qf[j], bf0, s0);
      s1 = wmma_f16(qf[j], bf1, s1);
    }

    // online softmax: a row lives across a 16-lane half (C/D layout)
    float alpha[8];
    for (int j = 0; j < 8; ++j) {
      float mx = fmaxf(s0[j], s1[j]);
      for (int d = 1; d < 16; d <<= 1) mx = fmaxf(mx, __shfl_xor(mx, d, 16));
      float nm = fmaxf(m[j], mx);
      float a  = __expf(m[j] - nm);
      float p0 = __expf(s0[j] - nm);
      float p1 = __expf(s1[j] - nm);
      float rs = p0 + p1;
      for (int d = 1; d < 16; d <<= 1) rs += __shfl_xor(rs, d, 16);
      l[j] = l[j] * a + rs;
      m[j] = nm; alpha[j] = a;
      s0[j] = p0; s1[j] = p1;
    }
    for (int ct = 0; ct < 8; ++ct)
      for (int j = 0; j < 8; ++j) acc[ct][j] *= alpha[j];

    // transpose P (C-layout) -> A-layout via per-wave LDS scratch
    _Float16* P = &sP[w * 16 * 32];
    {
      int rr = (lane & 16) ? 8 : 0, cc = lane & 15;
      for (int j = 0; j < 8; ++j) {
        P[(rr + j) * 32 + cc]      = (_Float16)s0[j];
        P[(rr + j) * 32 + 16 + cc] = (_Float16)s1[j];
      }
    }
    __syncthreads();

    v16h pf = frag16(&P[(lane & 15) * 32], lane);
    for (int ct = 0; ct < 8; ++ct) {      // O += P x V^T
      v16h vf = frag16(&bV[(ct * 16 + (lane & 15)) * 32], lane);
      acc[ct] = wmma_f16(pf, vf, acc[ct]);
    }
    __syncthreads();
  }

  // epilogue: O /= l, write f16 into hh[b'][q][768] at this instance's slot
  float inv[8];
  for (int j = 0; j < 8; ++j) inv[j] = 1.0f / l[j];
  const int rbase = qb * 128 + w * 16 + ((lane & 16) ? 8 : 0);
  _Float16* H = ws + OFF_HH + (size_t)bo * HW_ * FAN3;
  for (int ct = 0; ct < 8; ++ct) {
    int col = col768 + ct * 16 + (lane & 15);
    for (int j = 0; j < 8; ++j)
      H[(size_t)(rbase + j) * FAN3 + col] = (_Float16)(acc[ct][j] * inv[j]);
  }
}

// ---------------------------------------------------------------------------
// Kernel 5: final 1x1 conv + residual:
//   out[b][c][hw] = x[b][c][hw] + hh[b][hw][:] . W3[:,c] + b3[c]
// M=1024, K=768 (24 x 32), N=128, double-buffered async staging.
// ---------------------------------------------------------------------------
__global__ void k_out(const float* __restrict__ x, const float* __restrict__ b3,
                      const _Float16* __restrict__ ws, float* __restrict__ out) {
  const int mblk = blockIdx.x;            // 0..7
  const int b    = blockIdx.y;            // 0..15
  const int t = threadIdx.x, w = t >> 5, lane = t & 31;
  const _Float16* A  = ws + OFF_HH + (size_t)b * HW_ * FAN3;
  const _Float16* Wm = ws + OFF_W3;                      // [128][768]

  __shared__ alignas(16) _Float16 Asub[2][128 * 32];
  __shared__ alignas(16) _Float16 Bsub[2][128 * 32];

  const int m0 = mblk * 128;

  auto stage = [&](int buf, int k0) {
    for (int i = 0; i < 2; ++i) {
      int idx = t + 256 * i;
      int row = idx >> 2, part = (idx & 3) * 8;
      cp16_g2l(&Asub[buf][row * 32 + part], A  + (size_t)(m0 + row) * FAN3 + k0 + part);
      cp16_g2l(&Bsub[buf][row * 32 + part], Wm + (size_t)row * FAN3 + k0 + part);
    }
  };

  v8f acc[8];
  for (int i = 0; i < 8; ++i) acc[i] = v8f_zero();
  stage(0, 0);

  for (int ks = 0; ks < 24; ++ks) {
    wait_async();
    __syncthreads();
    if (ks + 1 < 24) stage((ks + 1) & 1, (ks + 1) * 32);
    const _Float16* As = Asub[ks & 1];
    const _Float16* Bs = Bsub[ks & 1];
    v16h af = frag16(&As[(w * 16 + (lane & 15)) * 32], lane);
    for (int nt = 0; nt < 8; ++nt) {
      v16h bf = frag16(&Bs[(nt * 16 + (lane & 15)) * 32], lane);
      acc[nt] = wmma_f16(af, bf, acc[nt]);
    }
    __syncthreads();
  }

  const int rbase = m0 + w * 16 + ((lane & 16) ? 8 : 0);
  for (int nt = 0; nt < 8; ++nt) {
    int c = nt * 16 + (lane & 15);
    float bv = b3[c];
    for (int j = 0; j < 8; ++j) {
      int row = rbase + j;
      size_t o = ((size_t)b * C_ + c) * HW_ + row;
      out[o] = acc[nt][j] + bv + x[o];
    }
  }
}

// ---------------------------------------------------------------------------
extern "C" void kernel_launch(void* const* d_in, const int* in_sizes, int n_in,
                              void* d_out, int out_size, void* d_ws, size_t ws_size,
                              hipStream_t stream) {
  const float* x        = (const float*)d_in[0];
  const float* q_cond   = (const float*)d_in[1];
  const float* k_a_cond = (const float*)d_in[2];
  const float* k_b_cond = (const float*)d_in[3];
  const float* gn_scale = (const float*)d_in[4];
  const float* gn_bias  = (const float*)d_in[5];
  const float* W0 = (const float*)d_in[6];
  const float* b0 = (const float*)d_in[7];
  const float* W1 = (const float*)d_in[8];
  const float* b1 = (const float*)d_in[9];
  const float* W2 = (const float*)d_in[10];
  const float* b2 = (const float*)d_in[11];
  const float* W3 = (const float*)d_in[12];
  const float* b3 = (const float*)d_in[13];
  float* out = (float*)d_out;
  _Float16* ws = (_Float16*)d_ws;

  k_wcvt <<<dim3((unsigned)((SZ_W3 + 255) / 256)), 256, 0, stream>>>(W0, W1, W2, W3, ws);
  k_cond <<<dim3(2048), 256, 0, stream>>>(q_cond, k_a_cond, k_b_cond, ws);
  k_gnorm<<<dim3(512),  256, 0, stream>>>(x, gn_scale, gn_bias, ws);
  k_qkv  <<<dim3(8, 4, 48), 256, 0, stream>>>(b0, b1, b2, ws);
  k_attn <<<dim3(8, 96),    256, 0, stream>>>(ws);
  k_out  <<<dim3(8, 16),    256, 0, stream>>>(x, b3, ws, out);
}
